// KDEAnomalyDetectorMean_31928786878792
// MI455X (gfx1250) — matrix operand order repaired and checked by reading, not compiled, not run on hardware
//
#include <hip/hip_runtime.h>
#include <hip/hip_bf16.h>

typedef __attribute__((ext_vector_type(16))) __bf16 v16bf;
typedef __attribute__((ext_vector_type(8)))  __bf16 bf16x8;
typedef __attribute__((ext_vector_type(8)))  float  v8f;

#define BATCH 8
#define NPTS  2048
#define DIM   256
#define EPS   1e-6f
#define THRESHOLD 2.0f

#define ROWS_PER_BLOCK 64
#define COLS_PER_ITER  32
#define LDS_STRIDE (DIM + 16)   // 272 bf16, multiple of 8 -> 16B aligned chunks
#define TILE_BYTES (COLS_PER_ITER * LDS_STRIDE * 2)          // 17408 B per col tile
#define SMEM_BYTES (ROWS_PER_BLOCK * LDS_STRIDE * 2)         // 34816 B = 2 col tiles

#if defined(__has_builtin)
#if __has_builtin(__builtin_amdgcn_global_load_async_to_lds_b128) && \
    __has_builtin(__builtin_amdgcn_s_wait_asynccnt)
#define USE_ASYNC_LDS 1
#endif
#endif
#ifndef USE_ASYNC_LDS
#define USE_ASYNC_LDS 0
#endif

#if USE_ASYNC_LDS
typedef int v4i __attribute__((vector_size(16)));           // matches builtin param type
typedef __attribute__((address_space(3))) v4i lds_v4i;      // LDS-side vector
#endif

// ---------------- Kernel 0: fp32 -> bf16 convert + per-row sum of squares ----
__global__ __launch_bounds__(256) void k_convert_sq(const float* __restrict__ x,
                                                    __bf16* __restrict__ xbf,
                                                    float* __restrict__ sq) {
    const int row = blockIdx.x;          // 0 .. B*N-1
    const int d   = threadIdx.x;         // 0 .. 255
    size_t i = (size_t)row * DIM + d;
    float v = x[i];
    xbf[i] = (__bf16)v;
    float s = v * v;
    for (int off = 1; off < 32; off <<= 1) s += __shfl_xor(s, off, 32);
    __shared__ float red[8];
    int lane = d & 31, w = d >> 5;
    if (lane == 0) red[w] = s;
    __syncthreads();
    if (d == 0) {
        float t = 0.f;
        for (int j = 0; j < 8; j++) t += red[j];
        sq[row] = t;
    }
}

// ---------------- Kernel 1: per-(b,d) std over N -----------------------------
__global__ __launch_bounds__(64) void k_col_stats(const float* __restrict__ x,
                                                  float* __restrict__ sigma) {
    const int b = blockIdx.x / DIM;
    const int d = blockIdx.x % DIM;
    const float* xb = x + (size_t)b * NPTS * DIM;
    float s = 0.f, s2 = 0.f;
    for (int n = threadIdx.x; n < NPTS; n += 64) {
        float v = xb[(size_t)n * DIM + d];
        s += v; s2 += v * v;
    }
    for (int off = 1; off < 32; off <<= 1) {
        s  += __shfl_xor(s,  off, 32);
        s2 += __shfl_xor(s2, off, 32);
    }
    __shared__ float rs[2], rs2[2];
    int w = threadIdx.x >> 5, lane = threadIdx.x & 31;
    if (lane == 0) { rs[w] = s; rs2[w] = s2; }
    __syncthreads();
    if (threadIdx.x == 0) {
        float S = rs[0] + rs[1], S2 = rs2[0] + rs2[1];
        float mean = S / (float)NPTS;
        float var  = fmaxf(S2 / (float)NPTS - mean * mean, 0.f); // ddof=0 (jnp.std default)
        sigma[blockIdx.x] = sqrtf(var);
    }
}

// ---------------- Kernel 2: bandwidth h per batch ----------------------------
__global__ __launch_bounds__(256) void k_bandwidth(const float* __restrict__ sigma,
                                                   float* __restrict__ h) {
    const int b = blockIdx.x;
    float v = sigma[b * DIM + threadIdx.x];
    for (int off = 1; off < 32; off <<= 1) v += __shfl_xor(v, off, 32);
    __shared__ float red[8];
    int lane = threadIdx.x & 31, w = threadIdx.x >> 5;
    if (lane == 0) red[w] = v;
    __syncthreads();
    if (threadIdx.x == 0) {
        float t = 0.f;
        for (int j = 0; j < 8; j++) t += red[j];
        float smean = t / (float)DIM + EPS;
        float coef = powf(4.0f / (float)(DIM + 2), 1.0f / (float)(DIM + 4)) *
                     powf((float)NPTS, -1.0f / (float)(DIM + 4));
        h[b] = coef * smean + EPS;
    }
}

// ---------------- Kernel 3: WMMA Gram + fused exp density --------------------
// 4 waves/block; each wave owns 16 rows with A fragments resident in VGPRs.
// Column tiles double-buffered in LDS; async DMA of tile i+1 overlaps the
// 16 WMMAs + exps of tile i (wait asynccnt<=8 instead of 0).
__global__ __launch_bounds__(128) void k_kde_density(const __bf16* __restrict__ xbf,
                                                     const float* __restrict__ sq,
                                                     const float* __restrict__ h,
                                                     float* __restrict__ densities) {
    const int blocksPerBatch = NPTS / ROWS_PER_BLOCK;
    const int b       = blockIdx.x / blocksPerBatch;
    const int rowBlk  = blockIdx.x % blocksPerBatch;
    const int rowBase0 = rowBlk * ROWS_PER_BLOCK;
    const int tid  = threadIdx.x;
    const int wave = tid >> 5;
    const int lane = tid & 31;
    const int l15  = lane & 15;
    const int halfSel = (lane < 16) ? 0 : 1;

    // one arena: prologue row staging, then ping/pong column buffers
    __shared__ __align__(16) char smem[SMEM_BYTES];
    __bf16* rowBuf  = (__bf16*)smem;
    __bf16* colBuf0 = (__bf16*)smem;
    __bf16* colBuf1 = (__bf16*)(smem + TILE_BYTES);

    const __bf16* xb  = xbf + (size_t)b * NPTS * DIM;
    const float*  sqb = sq  + (size_t)b * NPTS;

    // stage the 64-row tile into LDS (16B chunks)
    for (int idx = tid; idx < ROWS_PER_BLOCK * (DIM / 8); idx += 128) {
        int r  = idx / (DIM / 8);
        int kc = (idx % (DIM / 8)) * 8;
        *(bf16x8*)&rowBuf[r * LDS_STRIDE + kc] =
            *(const bf16x8*)&xb[(size_t)(rowBase0 + r) * DIM + kc];
    }
    __syncthreads();

    // hoist the wave's A fragments into registers (invariant over the m loop)
    // A layout (ISA): lanes 0-15 hold K 0-7,16-23; lanes 16-31 hold K 8-15,24-31
    const int khalf = halfSel * 8;
    v16bf aFrag[DIM / 32];
    for (int kt = 0; kt < DIM / 32; kt++) {
        const int k0 = kt * 32;
        bf16x8 alo = *(const bf16x8*)&rowBuf[(wave * 16 + l15) * LDS_STRIDE + k0 + khalf];
        bf16x8 ahi = *(const bf16x8*)&rowBuf[(wave * 16 + l15) * LDS_STRIDE + k0 + 16 + khalf];
        for (int i = 0; i < 8; i++) { aFrag[kt][i] = alo[i]; aFrag[kt][8 + i] = ahi[i]; }
    }
    __syncthreads();   // all waves done reading rowBuf before smem is reused

    // stage a 32-column tile (8 transfers per thread)
    auto stage_tile = [&](int m0, __bf16* dst) {
#if USE_ASYNC_LDS
        for (int idx = tid; idx < COLS_PER_ITER * (DIM / 8); idx += 128) {
            int r  = idx / (DIM / 8);
            int kc = (idx % (DIM / 8)) * 8;
            __builtin_amdgcn_global_load_async_to_lds_b128(
                (v4i*)&xb[(size_t)(m0 + r) * DIM + kc],
                (lds_v4i*)&dst[r * LDS_STRIDE + kc], 0, 0);
        }
#else
        for (int idx = tid; idx < COLS_PER_ITER * (DIM / 8); idx += 128) {
            int r  = idx / (DIM / 8);
            int kc = (idx % (DIM / 8)) * 8;
            *(bf16x8*)&dst[r * LDS_STRIDE + kc] =
                *(const bf16x8*)&xb[(size_t)(m0 + r) * DIM + kc];
        }
#endif
    };

    const float hv = h[b];
    const float invTwoH2 = 1.0f / (2.0f * hv * hv + EPS);

    const int myRowBase = rowBase0 + wave * 16;
    float sqr[8];
    for (int v = 0; v < 8; v++) sqr[v] = sqb[myRowBase + v + halfSel * 8];

    float densAcc[8];
    for (int v = 0; v < 8; v++) densAcc[v] = 0.f;

    const int NT = NPTS / COLS_PER_ITER;
    stage_tile(0, colBuf0);

    for (int it = 0; it < NT; ++it) {
        const int m0 = it * COLS_PER_ITER;
        __bf16* cur = (it & 1) ? colBuf1 : colBuf0;

        if (it + 1 < NT) {
            stage_tile(m0 + COLS_PER_ITER, (it & 1) ? colBuf0 : colBuf1);
#if USE_ASYNC_LDS
            __builtin_amdgcn_s_wait_asynccnt(8);   // tile `it` done (in-order); next tile in flight
#endif
        } else {
#if USE_ASYNC_LDS
            __builtin_amdgcn_s_wait_asynccnt(0);
#endif
        }
        __syncthreads();

        // two 16-col B tiles share the register-resident A fragments
        v8f c0 = {};
        v8f c1 = {};
        for (int kt = 0; kt < DIM / 32; kt++) {
            const int kb = kt * 32 + halfSel * 8;   // lanes<16: K 0-7,16-23; lanes>=16: K 8-15,24-31
            bf16x8 b0lo = *(const bf16x8*)&cur[l15 * LDS_STRIDE + kb];
            bf16x8 b0hi = *(const bf16x8*)&cur[l15 * LDS_STRIDE + kb + 16];
            bf16x8 b1lo = *(const bf16x8*)&cur[(16 + l15) * LDS_STRIDE + kb];
            bf16x8 b1hi = *(const bf16x8*)&cur[(16 + l15) * LDS_STRIDE + kb + 16];
            v16bf bb0, bb1;
            for (int i = 0; i < 8; i++) {
                bb0[i] = b0lo[i]; bb0[8 + i] = b0hi[i];
                bb1[i] = b1lo[i]; bb1[8 + i] = b1hi[i];
            }
            c0 = __builtin_amdgcn_wmma_f32_16x16x32_bf16(false, aFrag[kt], false, bb0,
                                                         (short)0, c0, false, false);
            c1 = __builtin_amdgcn_wmma_f32_16x16x32_bf16(false, aFrag[kt], false, bb1,
                                                         (short)0, c1, false, false);
        }
        // c[v] = dot(row myRowBase+v+8*halfSel, col); fuse distance + exp
        const float sqc0 = sqb[m0 + l15];
        const float sqc1 = sqb[m0 + 16 + l15];
        for (int v = 0; v < 8; v++) {
            float d0 = (sqr[v] + sqc0 - 2.0f * c0[v]) * (1.0f / (float)DIM);
            float d1 = (sqr[v] + sqc1 - 2.0f * c1[v]) * (1.0f / (float)DIM);
            densAcc[v] += __expf(-d0 * invTwoH2) + __expf(-d1 * invTwoH2);
        }
        __syncthreads();   // all reads of `cur` done before it is refilled
    }

    // reduce the 16 columns held across lanes of each half-wave
    for (int v = 0; v < 8; v++) {
        float s = densAcc[v];
        s += __shfl_xor(s, 1, 32);
        s += __shfl_xor(s, 2, 32);
        s += __shfl_xor(s, 4, 32);
        s += __shfl_xor(s, 8, 32);
        if (l15 == 0)
            densities[(size_t)b * NPTS + myRowBase + v + halfSel * 8] = s;
    }
}

// ---------------- Kernel 4: density mean/std(ddof=1), threshold, count -------
__global__ __launch_bounds__(256) void k_dens_stats(const float* __restrict__ densities,
                                                    float* __restrict__ thresh,
                                                    int* __restrict__ cnt) {
    const int b = blockIdx.x;
    const float* db = densities + (size_t)b * NPTS;
    float s = 0.f, s2 = 0.f;
    for (int n = threadIdx.x; n < NPTS; n += 256) {
        float v = db[n];
        s += v; s2 += v * v;
    }
    for (int off = 1; off < 32; off <<= 1) {
        s  += __shfl_xor(s,  off, 32);
        s2 += __shfl_xor(s2, off, 32);
    }
    __shared__ float rs[8], rs2[8];
    __shared__ float sh_thresh;
    int w = threadIdx.x >> 5, lane = threadIdx.x & 31;
    if (lane == 0) { rs[w] = s; rs2[w] = s2; }
    __syncthreads();
    if (threadIdx.x == 0) {
        float S = 0.f, S2 = 0.f;
        for (int j = 0; j < 8; j++) { S += rs[j]; S2 += rs2[j]; }
        float mean = S / (float)NPTS;
        float var  = fmaxf((S2 - (float)NPTS * mean * mean) / (float)(NPTS - 1), 0.f); // ddof=1
        sh_thresh = mean - THRESHOLD * sqrtf(var);
    }
    __syncthreads();
    float th = sh_thresh;
    int c = 0;
    for (int n = threadIdx.x; n < NPTS; n += 256) c += (db[n] > th) ? 1 : 0;
    for (int off = 1; off < 32; off <<= 1) c += __shfl_xor(c, off, 32);
    __shared__ int rc[8];
    if (lane == 0) rc[w] = c;
    __syncthreads();
    if (threadIdx.x == 0) {
        int C = 0;
        for (int j = 0; j < 8; j++) C += rc[j];
        thresh[b] = th;
        cnt[b] = C;
    }
}

// ---------------- Kernel 5: masked per-feature means -------------------------
__global__ __launch_bounds__(256) void k_masked_mean(const float* __restrict__ x,
                                                     const float* __restrict__ densities,
                                                     const float* __restrict__ thresh,
                                                     const int* __restrict__ cnt,
                                                     float* __restrict__ out) {
    const int b = blockIdx.x;
    const int d = threadIdx.x;
    const float* xb = x + (size_t)b * NPTS * DIM;
    const float* db = densities + (size_t)b * NPTS;
    const float th = thresh[b];
    float acc = 0.f;
    for (int n = 0; n < NPTS; n++) {
        if (db[n] > th) acc += xb[(size_t)n * DIM + d];   // coalesced across d
    }
    int c = cnt[b];
    int denom = (c < 1) ? 1 : c;
    out[b * DIM + d] = (c > 0) ? acc / (float)denom : 0.f;
}

extern "C" void kernel_launch(void* const* d_in, const int* in_sizes, int n_in,
                              void* d_out, int out_size, void* d_ws, size_t ws_size,
                              hipStream_t stream) {
    const float* x = (const float*)d_in[0];
    float* out = (float*)d_out;

    char* ws = (char*)d_ws;
    size_t off = 0;
    __bf16* xbf = (__bf16*)(ws + off);  off += (size_t)BATCH * NPTS * DIM * sizeof(__bf16); // 8 MB
    float* sq        = (float*)(ws + off); off += (size_t)BATCH * NPTS * sizeof(float);     // 64 KB
    float* densities = (float*)(ws + off); off += (size_t)BATCH * NPTS * sizeof(float);     // 64 KB
    float* sigma     = (float*)(ws + off); off += (size_t)BATCH * DIM * sizeof(float);      // 8 KB
    float* h         = (float*)(ws + off); off += 256;
    float* thresh    = (float*)(ws + off); off += 256;
    int*   cnt       = (int*)(ws + off);   off += 256;

    k_convert_sq <<<BATCH * NPTS, 256, 0, stream>>>(x, xbf, sq);
    k_col_stats  <<<BATCH * DIM, 64, 0, stream>>>(x, sigma);
    k_bandwidth  <<<BATCH, 256, 0, stream>>>(sigma, h);
    k_kde_density<<<BATCH * (NPTS / ROWS_PER_BLOCK), 128, 0, stream>>>(xbf, sq, h, densities);
    k_dens_stats <<<BATCH, 256, 0, stream>>>(densities, thresh, cnt);
    k_masked_mean<<<BATCH, 256, 0, stream>>>(x, densities, thresh, cnt, out);
}
